// HODLRLinear_34239479284209
// MI455X (gfx1250) — matrix operand order repaired and checked by reading, not compiled, not run on hardware
//
#include <hip/hip_runtime.h>
#include <hip/hip_bf16.h>

// ---------------------------------------------------------------------------
// HODLR linear for MI455X (gfx1250, wave32).
// Roofline: 34.4 GFLOP vs ~0.5 GB mandatory traffic -> bandwidth bound at
// 23.3 TB/s with bf16 WMMA (16x16x32, f32 accum). Read x once (LDS tile),
// write out once, weights preconverted to bf16 (2 MB, L2 resident).
// ---------------------------------------------------------------------------

typedef __attribute__((ext_vector_type(16))) __bf16 v16bf;
typedef __attribute__((ext_vector_type(8)))  __bf16 v8bf;
typedef __attribute__((ext_vector_type(4)))  __bf16 v4bf;
typedef __attribute__((ext_vector_type(8)))  float  v8f;

#define D        4096
#define TOK_TILE 16
#define NWAVES   8

__device__ __forceinline__ v16bf cat16(v8bf lo, v8bf hi) {
    union { v16bf v; v8bf h[2]; } u;
    u.h[0] = lo; u.h[1] = hi;
    return u.v;
}

__device__ __forceinline__ v8f wmma_bf16(v16bf a, v16bf b, v8f c) {
    // v_wmma_f32_16x16x32_bf16  D = A(16x32) * B(32x16) + C(16x16 f32)
    return __builtin_amdgcn_wmma_f32_16x16x32_bf16(
        /*neg_a=*/false, a, /*neg_b=*/false, b,
        /*c_mod=*/(short)0, c, /*reuse_a=*/false, /*reuse_b=*/false);
}

// cumulative block counts per level: n_i = 2^(i+1) -> 2,4,8,16,32,64
__device__ __forceinline__ constexpr int cumB(int i) {
    return (i == 0) ? 0 : (i == 1) ? 2 : (i == 2) ? 6 :
           (i == 3) ? 14 : (i == 4) ? 30 : 62;
}

// ---------------------------------------------------------------------------
// Prep: out[b][c][r] = (bf16) in[b][r][c]   (per-block transpose + downconvert)
// Puts every weight matrix in "output-index major, K contiguous" order so WMMA
// B fragments are 16B-contiguous per lane.
// ---------------------------------------------------------------------------
__global__ __launch_bounds__(256)
void transpose_to_bf16(const float* __restrict__ in, __bf16* __restrict__ out,
                       int rows, int cols, long total) {
    long idx    = (long)blockIdx.x * 256 + threadIdx.x;
    long stride = (long)gridDim.x * 256;
    long bc     = (long)rows * cols;
    for (; idx < total; idx += stride) {
        long b   = idx / bc;
        long rem = idx - b * bc;
        long c   = rem / rows;
        long r   = rem - c * rows;
        out[idx] = (__bf16)in[b * bc + r * cols + c];
    }
}

// ---------------------------------------------------------------------------
// Main fused kernel: one 16-token tile per workgroup (8 waves).
//   smem[0      .. 131071] : x tile, 16 x 4096 bf16 (row-major)
//   smem[131072 .. 195583] : t projections, 126 blocks x (16x16) bf16
// ---------------------------------------------------------------------------
__global__ __launch_bounds__(256)
void hodlr_main(const float*  __restrict__ x,
                const __bf16* __restrict__ usT,   // per level i: (n,16,w) K-contig
                const __bf16* __restrict__ vsT,   // per level i: (n,w,16) r-contig
                const __bf16* __restrict__ dsT,   // (64,64,64) outcol-major
                const float*  __restrict__ bias,
                float*        __restrict__ out)
{
    extern __shared__ char smem[];
    __bf16* xlds = (__bf16*)smem;                          // 16 x 4096
    __bf16* tlds = (__bf16*)(smem + TOK_TILE * D * 2);     // 126 x 256

    const int  tid  = threadIdx.x;
    const int  lane = tid & 31;
    // Force the wave index into an SGPR: all loop bounds / job decode below
    // become provably wave-uniform -> scalar branches, EXEC stays all-1s at
    // every WMMA (ISA requirement), no per-iteration exec masking.
    const int  wave = __builtin_amdgcn_readfirstlane(tid >> 5);
    const int  r15  = lane & 15;
    const int  hi   = lane >> 4;          // half-wave select (ISA lane layout)
    const long tt   = (long)blockIdx.x * TOK_TILE;

    // ---- stage x tile: f32 global -> bf16 LDS (coalesced float4 reads) ----
    {
        const float4* x4 = (const float4*)(x + tt * D);
        v4bf* l4 = (v4bf*)xlds;
        #pragma unroll
        for (int k = 0; k < (TOK_TILE * D / 4) / 256; ++k) {
            int idx = tid + 256 * k;
            float4 f = x4[idx];
            v4bf b;
            b[0] = (__bf16)f.x; b[1] = (__bf16)f.y;
            b[2] = (__bf16)f.z; b[3] = (__bf16)f.w;
            l4[idx] = b;
        }
    }
    __syncthreads();

    // =======================================================================
    // Phase 1: t[level][blk] (16x16) = x_blk (16 x w) @ us (w x 16)
    // 126 jobs round-robin across the 8 waves (uniform trip counts).
    // A fragment: lane<16 row=r15 holds K {0..7,16..23}; lane>=16 {8..15,24..31}
    // B fragment (K-contig layout): 8 bf16 at col*w + k0 + hi*8, 8 more at +16.
    // Two independent accumulator chains (even/odd 32-steps; every w is a
    // multiple of 64) so two load pairs are in flight per WMMA wait.
    // =======================================================================
    #pragma unroll 1
    for (int p = wave; p < 126; p += NWAVES) {
        int i, cum;
        if      (p < 2)  { i = 0; cum = 0;  }
        else if (p < 6)  { i = 1; cum = 2;  }
        else if (p < 14) { i = 2; cum = 6;  }
        else if (p < 30) { i = 3; cum = 14; }
        else if (p < 62) { i = 4; cum = 30; }
        else             { i = 5; cum = 62; }
        const int j = p - cum;
        const int w = D >> (i + 1);

        const __bf16* xrow = xlds + r15 * D + j * w + hi * 8;
        const __bf16* brow = usT + ((size_t)i << 16) + (size_t)j * (16 * w)
                                 + r15 * w + hi * 8;
        v8f acc0 = {}, acc1 = {};
        #pragma unroll 1
        for (int k0 = 0; k0 < w; k0 += 64) {
            __builtin_prefetch(brow + k0 + 128, 0, 3);  // global_prefetch_b8
            v16bf A0 = cat16(*(const v8bf*)(xrow + k0),
                             *(const v8bf*)(xrow + k0 + 16));
            v16bf B0 = cat16(*(const v8bf*)(brow + k0),
                             *(const v8bf*)(brow + k0 + 16));
            v16bf A1 = cat16(*(const v8bf*)(xrow + k0 + 32),
                             *(const v8bf*)(xrow + k0 + 48));
            v16bf B1 = cat16(*(const v8bf*)(brow + k0 + 32),
                             *(const v8bf*)(brow + k0 + 48));
            acc0 = wmma_bf16(A0, B0, acc0);
            acc1 = wmma_bf16(A1, B1, acc1);
        }
        v8f acc = acc0 + acc1;
        // C layout: VGPR v -> row (hi*8 + v), col r15.  Park as bf16 in LDS.
        __bf16* tp = tlds + (size_t)p * 256;
        #pragma unroll
        for (int v = 0; v < 8; ++v)
            tp[(hi * 8 + v) * 16 + r15] = (__bf16)acc[v];
    }
    __syncthreads();

    // =======================================================================
    // Phase 2: per 64-wide output column tile c:
    //   out(16x64) = [t cat over 6 levels (K=96)] @ [vs slices] (3 k-steps)
    //              + x_blk(16x64) @ ds[c] (2 k-steps) + bias
    // Four N-subtiles give 4 independent accumulator chains.
    // =======================================================================
    #pragma unroll 1
    for (int c = wave; c < 64; c += NWAVES) {
        v8f acc[4] = {{}, {}, {}, {}};

        // ---- low-rank contributions, K-concatenated across level pairs ----
        #pragma unroll
        for (int kp = 0; kp < 3; ++kp) {
            const int la = 2 * kp, lb = 2 * kp + 1;
            const int wa = D >> (la + 1), wb = D >> (lb + 1);
            const int ba = c >> (5 - la), bb = c >> (5 - lb); // out blocks
            const int ja = ba ^ 1,        jb = bb ^ 1;        // src blocks
            const int ca = c * 64 - ba * wa;  // col offset inside out block
            const int cb = c * 64 - bb * wb;

            // A: lo8 = t[la][ja] r 0..7|8..15 ; hi8 = t[lb][jb] same
            const __bf16* ta = tlds + (size_t)(cumB(la) + ja) * 256 + r15 * 16 + hi * 8;
            const __bf16* tb = tlds + (size_t)(cumB(lb) + jb) * 256 + r15 * 16 + hi * 8;
            v16bf A = cat16(*(const v8bf*)ta, *(const v8bf*)tb);

            const __bf16* va = vsT + ((size_t)la << 16) + (size_t)ja * (wa * 16);
            const __bf16* vb = vsT + ((size_t)lb << 16) + (size_t)jb * (wb * 16);
            #pragma unroll
            for (int s = 0; s < 4; ++s) {
                const int col = s * 16 + r15;
                // lo8 = vs[la] r hi*8..+7 ; hi8 = vs[lb] r hi*8..+7
                const __bf16* pa = va + (size_t)(ca + col) * 16 + hi * 8;
                const __bf16* pb = vb + (size_t)(cb + col) * 16 + hi * 8;
                v16bf B = cat16(*(const v8bf*)pa, *(const v8bf*)pb);
                acc[s] = wmma_bf16(A, B, acc[s]);
            }
        }

        // ---- block-diagonal 64x64 term ----
        {
            const __bf16* xrow = xlds + r15 * D + c * 64 + hi * 8;
            const __bf16* db   = dsT + (size_t)c * 4096 + hi * 8;
            #pragma unroll
            for (int k0 = 0; k0 < 64; k0 += 32) {
                v16bf A = cat16(*(const v8bf*)(xrow + k0),
                                *(const v8bf*)(xrow + k0 + 16));
                #pragma unroll
                for (int s = 0; s < 4; ++s) {
                    const __bf16* bp = db + (size_t)(s * 16 + r15) * 64 + k0;
                    v16bf B = cat16(*(const v8bf*)bp, *(const v8bf*)(bp + 16));
                    acc[s] = wmma_bf16(A, B, acc[s]);
                }
            }
        }

        // ---- bias + store (C layout: VGPR v -> row hi*8+v, col r15) ----
        #pragma unroll
        for (int s = 0; s < 4; ++s) {
            const int g  = c * 64 + s * 16 + r15;
            const float bv = bias[g];
            float* op = out + (tt + hi * 8) * (long)D + g;
            #pragma unroll
            for (int v = 0; v < 8; ++v)
                op[(size_t)v * D] = acc[s][v] + bv;
        }
    }
}

// ---------------------------------------------------------------------------
// Host-side launch. d_in order (setup_inputs dict order):
//   [0]=x, [1]=us0,[2]=vs0, [3]=us1,[4]=vs1, ... [11]=us5,[12]=vs5,
//   [13]=ds, [14]=bias
// Workspace layout (bytes):
//   [0        , 786432 ) usT  : 6 levels x 65536 bf16
//   [786432   , 1572864) vsT  : 6 levels x 65536 bf16
//   [1572864  , 2097152) dsT  : 262144 bf16
// ---------------------------------------------------------------------------
extern "C" void kernel_launch(void* const* d_in, const int* in_sizes, int n_in,
                              void* d_out, int out_size, void* d_ws, size_t ws_size,
                              hipStream_t stream) {
    (void)in_sizes; (void)n_in; (void)out_size; (void)ws_size;

    const float* x    = (const float*)d_in[0];
    const float* ds   = (const float*)d_in[13];
    const float* bias = (const float*)d_in[14];
    float*       out  = (float*)d_out;

    char* ws = (char*)d_ws;
    __bf16* usT = (__bf16*)(ws);
    __bf16* vsT = (__bf16*)(ws + 786432);
    __bf16* dsT = (__bf16*)(ws + 1572864);

    // Weight prep: transpose to K-contiguous bf16 (tiny: ~2 MB total).
    for (int i = 0; i < 6; ++i) {
        const float* usi = (const float*)d_in[1 + 2 * i];
        const float* vsi = (const float*)d_in[2 + 2 * i];
        const int n = 2 << i;
        const int w = D / n;
        const long tot = (long)n * w * 16;
        // usT[b][r][k] = us[b][k][r]   (rows=w, cols=16)
        transpose_to_bf16<<<64, 256, 0, stream>>>(usi, usT + (size_t)i * 65536,
                                                  w, 16, tot);
        // vsT[b][col][r] = vs[b][r][col] (rows=16, cols=w)
        transpose_to_bf16<<<64, 256, 0, stream>>>(vsi, vsT + (size_t)i * 65536,
                                                  16, w, tot);
    }
    // dsT[b][outc][k] = ds[b][k][outc]
    transpose_to_bf16<<<256, 256, 0, stream>>>(ds, dsT, 64, 64, 64L * 64 * 64);

    // Main fused kernel: 16384 tokens / 16 per tile = 1024 workgroups.
    const size_t lds_bytes = TOK_TILE * D * 2 + 126 * 256 * 2;  // 195584 B
    hodlr_main<<<1024, 256, lds_bytes, stream>>>(x, usT, vsT, dsT, bias, out);
}
